// LowRankAttention_30872224923817
// MI455X (gfx1250) — matrix operand order repaired and checked by reading, not compiled, not run on hardware
//
#include <hip/hip_runtime.h>
#include <hip/hip_bf16.h>

typedef __attribute__((ext_vector_type(16))) __bf16 v16bf;
typedef __attribute__((ext_vector_type(8)))  __bf16 bf16x8;
typedef __attribute__((ext_vector_type(8)))  float  v8f;
typedef __attribute__((ext_vector_type(4)))  int    v4i;

#define DEV __device__ __forceinline__

// ---- CDNA5 async global->LDS (ASYNCcnt) with safe fallback ----------------
#if defined(__AMDGCN__) && __has_builtin(__builtin_amdgcn_global_load_async_to_lds_b128)
#define USE_ASYNC_LDS 1
typedef __attribute__((address_space(1))) v4i* gv4i_p;
typedef __attribute__((address_space(3))) v4i* lv4i_p;
DEV void async_copy16(const void* g, void* l) {
    __builtin_amdgcn_global_load_async_to_lds_b128((gv4i_p)(g), (lv4i_p)(l), 0, 0);
}
template <int N> DEV void wait_async() {
#if __has_builtin(__builtin_amdgcn_s_wait_asynccnt)
    __builtin_amdgcn_s_wait_asynccnt(N);
#else
    asm volatile("s_wait_asynccnt %0" :: "i"(N) : "memory");
#endif
}
#endif

// ---- WMMA fragment helpers (bf16, 16x16x32) -------------------------------
// A (16x32) layout: lane<16 -> M=lane, halves0-15:K0-15
//                   lane>=16 -> M=lane-16, halves0-15:K16-31
// B (32x16) layout: n = lane&15, kbase=(lane>>4)*16, half h -> K=kbase+h
//   -> keep B K-contiguous (n-major) so frag loads are b128.
// C/D layout: VGPR r, lane l -> M = r + 8*(l>>4), N = l&15.
DEV v16bf frag_join(bf16x8 lo, bf16x8 hi) {
    v16bf f;
#pragma unroll
    for (int i = 0; i < 8; ++i) { f[i] = lo[i]; f[i + 8] = hi[i]; }
    return f;
}
DEV v16bf load_fragA(const __bf16* p, int lane) {
    int khi = (lane >> 4) * 16;
    bf16x8 lo = *(const bf16x8*)(p + khi);
    bf16x8 hi = *(const bf16x8*)(p + 8 + khi);
    return frag_join(lo, hi);
}
DEV v16bf load_fragB(const __bf16* p) {
    bf16x8 lo = *(const bf16x8*)(p);
    bf16x8 hi = *(const bf16x8*)(p + 8);
    return frag_join(lo, hi);
}
DEV v8f wmma_bf16(v16bf a, v16bf b, v8f c) {
    return __builtin_amdgcn_wmma_f32_16x16x32_bf16(false, a, false, b, (short)0, c, false, false);
}

// ---- one-time conversions -------------------------------------------------
__global__ __launch_bounds__(256)
void cvt_f32_bf16_kernel(const float* __restrict__ in, __bf16* __restrict__ out, int n8) {
    int i = blockIdx.x * 256 + threadIdx.x;
    if (i >= n8) return;
    size_t base = (size_t)i * 8;
    bf16x8 o;
#pragma unroll
    for (int j = 0; j < 8; ++j) o[j] = (__bf16)in[base + j];
    *(bf16x8*)(out + base) = o;
}

// wt[n][k] = (bf16) w[k][n]   (each thread: fixed n, 8 consecutive k)
__global__ __launch_bounds__(256)
void transpose_cvt_kernel(const float* __restrict__ w, __bf16* __restrict__ wt, int K, int N) {
    int i = blockIdx.x * 256 + threadIdx.x;
    int kpr = K >> 3;
    int n = i / kpr, k0 = (i % kpr) * 8;
    if (n >= N) return;
    bf16x8 o;
#pragma unroll
    for (int j = 0; j < 8; ++j) o[j] = (__bf16)w[(size_t)(k0 + j) * N + n];
    *(bf16x8*)(wt + (size_t)n * K + k0) = o;
}

// ---- Tiled GEMM: A[M,K] bf16 x Bt[N,K] bf16 (+bias) -----------------------
// Block = 256 thr = 8 waves, tile 128x128, k-step 64, double-buffered LDS,
// async global->LDS staging when available.
template <bool OUT_BF16, bool ADD_BIAS>
__global__ __launch_bounds__(256)
void gemm_bf16_kernel(const __bf16* __restrict__ A, const __bf16* __restrict__ Bt,
                      const float* __restrict__ bias, void* __restrict__ Cp,
                      int M, int N, int K) {
    __shared__ __bf16 As[2][128][64];
    __shared__ __bf16 Bs[2][128][64];
    const int tid  = threadIdx.x;
    const int wave = tid >> 5, lane = tid & 31;
    const int lm = lane & 15, lg = lane >> 4;
    const int bm = blockIdx.y * 128, bn = blockIdx.x * 128;
    const int sr = tid >> 1;           // staging row 0..127
    const int sc = (tid & 1) * 32;     // staging col base 0/32

    v8f acc[8];
#pragma unroll
    for (int t = 0; t < 8; ++t) acc[t] = (v8f){0.f,0.f,0.f,0.f,0.f,0.f,0.f,0.f};

    auto stage = [&](int buf, int k0) {
        const __bf16* ga = A  + (size_t)(bm + sr) * K + k0 + sc;
        const __bf16* gb = Bt + (size_t)(bn + sr) * K + k0 + sc;
#ifdef USE_ASYNC_LDS
#pragma unroll
        for (int c2 = 0; c2 < 4; ++c2) {
            async_copy16(ga + c2 * 8, &As[buf][sr][sc + c2 * 8]);
            async_copy16(gb + c2 * 8, &Bs[buf][sr][sc + c2 * 8]);
        }
#else
#pragma unroll
        for (int c2 = 0; c2 < 4; ++c2) {
            *(bf16x8*)&As[buf][sr][sc + c2 * 8] = *(const bf16x8*)(ga + c2 * 8);
            *(bf16x8*)&Bs[buf][sr][sc + c2 * 8] = *(const bf16x8*)(gb + c2 * 8);
        }
#endif
    };

    const int nsteps = K >> 6;
    stage(0, 0);
    int buf = 0;
    for (int s = 0; s < nsteps; ++s) {
        const bool has_next = (s + 1) < nsteps;
        if (has_next) stage(buf ^ 1, (s + 1) << 6);   // overlap with compute below
#ifdef USE_ASYNC_LDS
        if (has_next) wait_async<8>(); else wait_async<0>();  // retire current stage
#endif
        __syncthreads();
        v16bf af0 = load_fragA(&As[buf][wave * 16 + lm][0],  lane);
        v16bf af1 = load_fragA(&As[buf][wave * 16 + lm][32], lane);
#pragma unroll
        for (int t = 0; t < 8; ++t) {
            const __bf16* bp = &Bs[buf][t * 16 + lm][lg * 16];
            acc[t] = wmma_bf16(af0, load_fragB(bp), acc[t]);
            acc[t] = wmma_bf16(af1, load_fragB(bp + 32), acc[t]);
        }
        __syncthreads();   // reads done before this buffer is overwritten
        buf ^= 1;
    }

    const int r0 = bm + wave * 16 + lg * 8;
    const int c0 = bn + lm;
#pragma unroll
    for (int t = 0; t < 8; ++t) {
        int col = c0 + t * 16;
        float bv = ADD_BIAS ? bias[col] : 0.f;
#pragma unroll
        for (int r = 0; r < 8; ++r) {
            float v = acc[t][r] + bv;
            size_t idx = (size_t)(r0 + r) * N + col;
            if (OUT_BF16) ((__bf16*)Cp)[idx] = (__bf16)v;
            else          ((float*)Cp)[idx]  = v;
        }
    }
}

// ---- Spatial attention: one block per (b,h,c); L=256 keys, hd=64 ----------
// qkv layout: [32768][4608] bf16, row m=((b*256+n)*16+c), col=(s*12+h)*64+d
__global__ __launch_bounds__(256)
void attn_spatial_kernel(const __bf16* __restrict__ qkv, float* __restrict__ xs) {
    extern __shared__ char smem[];
    __bf16* Klds = (__bf16*)smem;           // [256][64] row-major (d contiguous)
    __bf16* Vt   = Klds + 256 * 64;         // [64][256] : Vt[d][n] (key-dim contiguous)
    __bf16* Pall = Vt + 64 * 256;           // [8 waves][16][256]

    const int tid = threadIdx.x, wave = tid >> 5, lane = tid & 31;
    const int c = blockIdx.x & 15;
    const int h = (blockIdx.x >> 4) % 12;
    const int b = blockIdx.x / (16 * 12);
    const float scale = 0.125f;  // 1/sqrt(64)

    {
        const int n = tid;
        const size_t rowm = ((size_t)(b * 256 + n) * 16 + c) * 4608;
        const __bf16* kr = qkv + rowm + (size_t)(36 + h) * 64;  // ks (s=3)
        const __bf16* vr = qkv + rowm + (size_t)(60 + h) * 64;  // vs (s=5)
#pragma unroll
        for (int d = 0; d < 64; d += 8)
            *(bf16x8*)&Klds[n * 64 + d] = *(const bf16x8*)&kr[d];
#pragma unroll
        for (int d = 0; d < 64; ++d) Vt[d * 256 + n] = vr[d];
    }
    __syncthreads();

    __bf16* pw = Pall + wave * 16 * 256;

#pragma unroll
    for (int it = 0; it < 2; ++it) {
        const int qt = wave * 2 + it;
        const int qrow = qt * 16 + (lane & 15);
        const __bf16* qr = qkv + ((size_t)(b * 256 + qrow) * 16 + c) * 4608 + (size_t)(24 + h) * 64; // qs
        v16bf aq0 = load_fragA(qr, lane);
        v16bf aq1 = load_fragA(qr + 32, lane);

        v8f st[16];
#pragma unroll
        for (int j = 0; j < 16; ++j) {
            v8f s = (v8f){0.f,0.f,0.f,0.f,0.f,0.f,0.f,0.f};
            const __bf16* kr = &Klds[(j * 16 + (lane & 15)) * 64 + (lane >> 4) * 16];
            s = wmma_bf16(aq0, load_fragB(kr), s);
            s = wmma_bf16(aq1, load_fragB(kr + 32), s);
            st[j] = s;
        }

        float rmax[8], rsum[8];
#pragma unroll
        for (int r = 0; r < 8; ++r) {
            float m = -3.0e38f;
#pragma unroll
            for (int j = 0; j < 16; ++j) m = fmaxf(m, st[j][r]);
#pragma unroll
            for (int msk = 1; msk < 16; msk <<= 1) m = fmaxf(m, __shfl_xor(m, msk, 32));
            rmax[r] = m;
        }
#pragma unroll
        for (int r = 0; r < 8; ++r) {
            float sum = 0.f;
#pragma unroll
            for (int j = 0; j < 16; ++j) {
                float e = __expf((st[j][r] - rmax[r]) * scale);
                st[j][r] = e;
                sum += e;
            }
#pragma unroll
            for (int msk = 1; msk < 16; msk <<= 1) sum += __shfl_xor(sum, msk, 32);
            rsum[r] = sum;
        }

#pragma unroll
        for (int j = 0; j < 16; ++j) {
            int colp = j * 16 + (lane & 15);
#pragma unroll
            for (int r = 0; r < 8; ++r)
                pw[(r + 8 * (lane >> 4)) * 256 + colp] = (__bf16)st[j][r];
        }

#pragma unroll
        for (int nt = 0; nt < 4; ++nt) {
            v8f o = (v8f){0.f,0.f,0.f,0.f,0.f,0.f,0.f,0.f};
#pragma unroll
            for (int kc = 0; kc < 8; ++kc) {
                v16bf ap = load_fragA(pw + (lane & 15) * 256 + kc * 32, lane);
                v16bf bv = load_fragB(Vt + (nt * 16 + (lane & 15)) * 256 + kc * 32 + (lane >> 4) * 16);
                o = wmma_bf16(ap, bv, o);
            }
            const int d = nt * 16 + (lane & 15);
#pragma unroll
            for (int r = 0; r < 8; ++r) {
                int n = qt * 16 + r + 8 * (lane >> 4);
                xs[(((size_t)(b * 12 + h) * 256 + n) * 16 + c) * 64 + d] = o[r] / rsum[r];
            }
        }
    }
}

// ---- Channel attention (C=16) + combine xs*xc -> bf16 [32768][768] --------
__global__ __launch_bounds__(256)
void attn_channel_kernel(const __bf16* __restrict__ qkv, const float* __restrict__ xs,
                         __bf16* __restrict__ comb) {
    __shared__ __bf16 Plds[8][16][32];
    const int tid = threadIdx.x, wave = tid >> 5, lane = tid & 31;
    const int idx = blockIdx.x * 8 + wave;
    const int n = idx & 255;
    const int h = (idx >> 8) % 12;
    const int b = idx / (256 * 12);
    const float scale = 0.125f;

    const size_t rowb = (size_t)(b * 256 + n) * 16 * 4608;
    const size_t cq = (size_t)h * 64, ck = (size_t)(12 + h) * 64, cv = (size_t)(48 + h) * 64;

    const __bf16* qr = qkv + rowb + (size_t)(lane & 15) * 4608 + cq;
    const __bf16* kr = qkv + rowb + (size_t)(lane & 15) * 4608 + ck + (lane >> 4) * 16;
    v8f s = (v8f){0.f,0.f,0.f,0.f,0.f,0.f,0.f,0.f};
    s = wmma_bf16(load_fragA(qr, lane), load_fragB(kr), s);
    s = wmma_bf16(load_fragA(qr + 32, lane), load_fragB(kr + 32), s);

    float rsum[8];
#pragma unroll
    for (int r = 0; r < 8; ++r) {
        float m = s[r];
#pragma unroll
        for (int msk = 1; msk < 16; msk <<= 1) m = fmaxf(m, __shfl_xor(m, msk, 32));
        float e = __expf((s[r] - m) * scale);
        float sum = e;
#pragma unroll
        for (int msk = 1; msk < 16; msk <<= 1) sum += __shfl_xor(sum, msk, 32);
        s[r] = e;
        rsum[r] = sum;
    }

#pragma unroll
    for (int i = 0; i < 16; ++i) Plds[wave][lane & 15][(lane >> 4) * 16 + i] = (__bf16)0.f;
#pragma unroll
    for (int r = 0; r < 8; ++r) Plds[wave][r + 8 * (lane >> 4)][lane & 15] = (__bf16)s[r];

    v16bf ap = load_fragA(&Plds[wave][lane & 15][0], lane);

#pragma unroll
    for (int nt = 0; nt < 4; ++nt) {
        const int d = nt * 16 + (lane & 15);
        v16bf bv;
        if (lane < 16) {
#pragma unroll
            for (int cc = 0; cc < 16; ++cc)
                bv[cc] = qkv[rowb + (size_t)cc * 4608 + cv + d];
        } else {
#pragma unroll
            for (int cc = 0; cc < 16; ++cc) bv[cc] = (__bf16)0.f;
        }
        v8f o = (v8f){0.f,0.f,0.f,0.f,0.f,0.f,0.f,0.f};
        o = wmma_bf16(ap, bv, o);
#pragma unroll
        for (int r = 0; r < 8; ++r) {
            int c = r + 8 * (lane >> 4);
            float xc = o[r] / rsum[r];
            float xv = xs[(((size_t)(b * 12 + h) * 256 + n) * 16 + c) * 64 + d];
            comb[((size_t)(b * 256 + n) * 16 + c) * 768 + (size_t)h * 64 + d] = (__bf16)(xc * xv);
        }
    }
}

extern "C" void kernel_launch(void* const* d_in, const int* in_sizes, int n_in,
                              void* d_out, int out_size, void* d_ws, size_t ws_size,
                              hipStream_t stream) {
    const float* x      = (const float*)d_in[0];   // [32768, 768]
    const float* w_qkv  = (const float*)d_in[1];   // [768, 4608]
    const float* w_proj = (const float*)d_in[2];   // [768, 768]
    const float* b_proj = (const float*)d_in[3];   // [768]
    float* out = (float*)d_out;

    const int M = 8 * 256 * 16, D = 768, N6 = 6 * 768;

    // workspace carve-up
    __bf16* qkv  = (__bf16*)d_ws;                               // M*4608 bf16
    float*  xs   = (float*)(qkv + (size_t)M * N6);              // 8*12*256*16*64 f32
    __bf16* comb = (__bf16*)(xs + (size_t)8 * 12 * 256 * 1024); // M*768 bf16
    __bf16* xb   = comb + (size_t)M * D;                        // M*768 bf16
    __bf16* wqt  = xb + (size_t)M * D;                          // [4608][768] bf16
    __bf16* wpt  = wqt + (size_t)N6 * D;                        // [768][768] bf16

    // 0) one-time conversions: x -> bf16; weights -> bf16 transposed [N][K]
    cvt_f32_bf16_kernel<<<(M * D / 8 + 255) / 256, 256, 0, stream>>>(x, xb, M * D / 8);
    transpose_cvt_kernel<<<(N6 * (D / 8) + 255) / 256, 256, 0, stream>>>(w_qkv, wqt, D, N6);
    transpose_cvt_kernel<<<(D * (D / 8) + 255) / 256, 256, 0, stream>>>(w_proj, wpt, D, D);

    // 1) QKV projection (bf16 x bf16 -> bf16), async double-buffered staging
    gemm_bf16_kernel<true, false>
        <<<dim3(N6 / 128, M / 128), 256, 0, stream>>>(xb, wqt, nullptr, qkv, M, N6, D);

    // 2) Spatial attention over N (one block per (b,h,c)); 128 KB dynamic LDS
    attn_spatial_kernel<<<8 * 12 * 16, 256, 131072, stream>>>(qkv, xs);

    // 3) Channel attention over C + elementwise combine (8 problems/block)
    attn_channel_kernel<<<(8 * 12 * 256) / 8, 256, 0, stream>>>(qkv, xs, comb);

    // 4) Output projection (bf16 A, +bias, f32 out)
    gemm_bf16_kernel<false, true>
        <<<dim3(D / 128, M / 128), 256, 0, stream>>>(comb, wpt, b_proj, out, M, D, D);
}
